// transformerReg_PAM_3934190044190
// MI455X (gfx1250) — matrix operand order repaired and checked by reading, not compiled, not run on hardware
//
#include <hip/hip_runtime.h>
#include <hip/hip_bf16.h>

// ---------------------------------------------------------------------------
// Problem constants
// ---------------------------------------------------------------------------
#define NPTS   32768
#define CFEAT  61
#define DDIM   64          // hidden width (GEMM N and K)
#define DLAT   8           // lattice dimension d
#define NCORN  9           // d+1 simplex corners
#define MCAP   (NPTS * NCORN)   // max distinct lattice entries = 294912

// Hash table
#define HBITS  20
#define HSIZE  (1u << HBITS)
#define HMASK  (HSIZE - 1u)
#define EMPTY64 0xFFFFFFFFFFFFFFFFull

// hash multipliers from the reference (int64, reinterpreted as wrapping u64)
constexpr unsigned long long kMults[8] = {
    2654435761405764093ull, 1181783497276652981ull, 3202034522624059733ull,
    2685821657736338717ull, 1876998521354586173ull, 1481765933965188213ull,
    2549297995355413921ull, 3373259426345127233ull};

typedef __attribute__((ext_vector_type(16))) _Float16 v16h;
typedef __attribute__((ext_vector_type(8)))  _Float16 v8h;
typedef __attribute__((ext_vector_type(8)))  float    v8f;

// ---------------------------------------------------------------------------
// Fused 2-layer MLP:  Y = relu(relu(X @ W0 + b0) @ W1 + b1)
// X, Y: NPTS x 64 f32.  One wave owns a 16-row tile end-to-end:
//   - layer-0 WMMAs from global X (A frag) x LDS-packed W0 (B frag)
//   - 16x64 f16 intermediate staged in per-wave LDS (row major)
//   - re-read in A-fragment layout, layer-1 WMMAs, bias+relu, store to HBM.
// Weights are pre-swizzled in LDS into B-fragment lane order so every
// fragment load is one contiguous 32-byte LDS read (2x ds_load_b128).
// Biases are fetched via CDNA5 async global->LDS loads (ASYNCcnt).
// 8 waves / block (256 thr), 256 blocks.
// ---------------------------------------------------------------------------
__global__ __launch_bounds__(256)
void fused_mlp_wmma(const float* __restrict__ X,
                    const float* __restrict__ W0, const float* __restrict__ b0,
                    const float* __restrict__ W1, const float* __restrict__ b1,
                    float* __restrict__ Y) {
  // packed B fragments: index = ((t*2+s)*32 + lane)*16 + i
  //   (t: 16-col tile, s: 32-deep k step, i: half within lane)
  __shared__ __align__(32) _Float16 Wp0[4096];
  __shared__ __align__(32) _Float16 Wp1[4096];
  __shared__ __align__(16) _Float16 stage[8][16 * DDIM];  // per-wave 16x64 f16
  __shared__ float bsh[128];                              // b0 | b1

  const int tid = threadIdx.x;

  // async global->LDS bias fetch (ISA 10.x GLOBAL_LOAD_ASYNC_TO_LDS_B32)
  if (tid < DDIM) {
    unsigned l0 = (unsigned)(unsigned long long)(const void*)&bsh[tid];
    unsigned l1 = (unsigned)(unsigned long long)(const void*)&bsh[DDIM + tid];
    const float* g0 = b0 + tid;
    const float* g1 = b1 + tid;
    asm volatile("global_load_async_to_lds_b32 %0, %1, off"
                 :: "v"(l0), "v"(g0) : "memory");
    asm volatile("global_load_async_to_lds_b32 %0, %1, off"
                 :: "v"(l1), "v"(g1) : "memory");
    asm volatile("s_wait_asynccnt 0" ::: "memory");
  }

  // pack both weight matrices into B-fragment lane order (f16)
  for (int p = tid; p < 4096; p += 256) {
    const int f  = p >> 9;          // fragment 0..7  (t*2+s)
    const int r  = p & 511;
    const int ln = r >> 4;          // lane 0..31
    const int i  = r & 15;          // half 0..15
    const int t  = f >> 1;
    const int s  = f & 1;
    const int K  = s * 32 + 8 * (ln >> 4) + (i < 8 ? i : i + 8);
    const int N  = t * 16 + (ln & 15);
    Wp0[p] = (_Float16)W0[K * DDIM + N];
    Wp1[p] = (_Float16)W1[K * DDIM + N];
  }
  __syncthreads();

  const int wave = tid >> 5;
  const int lane = tid & 31;
  const int H    = lane >> 4;
  const int l16  = lane & 15;

  const long long rowBase = (long long)(blockIdx.x * 8 + wave) * 16;
  const float* Xrow = X + (rowBase + l16) * DDIM;
  __builtin_prefetch(Xrow, 0, 1);   // -> global_prefetch_b8

  // ---- layer 0: A fragments straight from global X -----------------------
  // A layout (ISA 7.12.2): lane half i -> K = s*32 + 8*H + (i<8 ? i : i+8)
  v16h a[2];
#pragma unroll
  for (int s = 0; s < 2; ++s) {
    const int kb = s * 32 + 8 * H;
#pragma unroll
    for (int i = 0; i < 8; ++i) a[s][i]     = (_Float16)Xrow[kb + i];
#pragma unroll
    for (int i = 0; i < 8; ++i) a[s][8 + i] = (_Float16)Xrow[kb + 16 + i];
  }

  _Float16* st = &stage[wave][0];
#pragma unroll
  for (int t = 0; t < 4; ++t) {
    const int n = t * 16 + l16;
    v8f acc = {};   // inline-0 C operand on first WMMA
#pragma unroll
    for (int s = 0; s < 2; ++s) {
      const v16h bfrag = *(const v16h*)&Wp0[(((t << 1) | s) << 9) + (lane << 4)];
      acc = __builtin_amdgcn_wmma_f32_16x16x32_f16(false, a[s], false, bfrag,
                                                   (short)0, acc, false, false);
    }
    const float bv = bsh[n];
#pragma unroll
    for (int v = 0; v < 8; ++v) {
      float r = acc[v] + bv;
      r = r > 0.f ? r : 0.f;
      // C/D element (M = v + 8*H, N = n) -> row-major staging
      st[(v + 8 * H) * DDIM + n] = (_Float16)r;
    }
  }
  __syncthreads();   // cross-lane LDS visibility (same wave suffices, be safe)

  // ---- layer 1: A fragments re-read from LDS stage -----------------------
  v16h a2[2];
#pragma unroll
  for (int s = 0; s < 2; ++s) {
    const int o1 = l16 * DDIM + s * 32 + 8 * H;      // 8 contiguous halves
    const v8h lo = *(const v8h*)&st[o1];
    const v8h hi = *(const v8h*)&st[o1 + 16];
#pragma unroll
    for (int i = 0; i < 8; ++i) { a2[s][i] = lo[i]; a2[s][8 + i] = hi[i]; }
  }

#pragma unroll
  for (int t = 0; t < 4; ++t) {
    const int n = t * 16 + l16;
    v8f acc = {};
#pragma unroll
    for (int s = 0; s < 2; ++s) {
      const v16h bfrag = *(const v16h*)&Wp1[(((t << 1) | s) << 9) + (lane << 4)];
      acc = __builtin_amdgcn_wmma_f32_16x16x32_f16(false, a2[s], false, bfrag,
                                                   (short)0, acc, false, false);
    }
    const float bv = bsh[DDIM + n];
#pragma unroll
    for (int v = 0; v < 8; ++v) {
      float r = acc[v] + bv;
      r = r > 0.f ? r : 0.f;
      Y[(rowBase + v + 8 * H) * DDIM + n] = r;
    }
  }
}

// ---------------------------------------------------------------------------
// pos GEMM: praw = relu(concat(kpts,feats) @ Wf + bf)  (NPTS x 8),
// plus global sum / sum-of-squares reduction for layernorm2.
// ---------------------------------------------------------------------------
__global__ __launch_bounds__(256)
void pos_gemm_reduce(const float* __restrict__ kpts, const float* __restrict__ feats,
                     const float* __restrict__ Wf, const float* __restrict__ bf,
                     float* __restrict__ praw, float* __restrict__ red) {
  __shared__ float Wl[DDIM * DLAT];
  __shared__ float bl[DLAT];
  __shared__ float rsum[256], rsq[256];
  const int tid = threadIdx.x;
  for (int i = tid; i < DDIM * DLAT; i += 256) Wl[i] = Wf[i];
  if (tid < DLAT) bl[tid] = bf[tid];
  __syncthreads();

  const long long i = (long long)blockIdx.x * 256 + tid;  // point index
  float acc[DLAT];
#pragma unroll
  for (int j = 0; j < DLAT; ++j) acc[j] = bl[j];
#pragma unroll
  for (int k = 0; k < 3; ++k) {
    const float x = kpts[i * 3 + k];
#pragma unroll
    for (int j = 0; j < DLAT; ++j) acc[j] = fmaf(x, Wl[k * DLAT + j], acc[j]);
  }
  for (int k = 0; k < CFEAT; ++k) {
    const float x = feats[i * CFEAT + k];
#pragma unroll
    for (int j = 0; j < DLAT; ++j) acc[j] = fmaf(x, Wl[(3 + k) * DLAT + j], acc[j]);
  }
  float s = 0.f, q = 0.f;
#pragma unroll
  for (int j = 0; j < DLAT; ++j) {
    float r = acc[j] > 0.f ? acc[j] : 0.f;
    praw[i * DLAT + j] = r;
    s += r;
    q += r * r;
  }
  rsum[tid] = s; rsq[tid] = q;
  __syncthreads();
  for (int st = 128; st > 0; st >>= 1) {
    if (tid < st) { rsum[tid] += rsum[tid + st]; rsq[tid] += rsq[tid + st]; }
    __syncthreads();
  }
  if (tid == 0) { atomicAdd(&red[0], rsum[0]); atomicAdd(&red[1], rsq[0]); }
}

// pos = (praw - mu) * rsqrt(var + 1e-5) * g + be   (global mean/var, in place)
__global__ __launch_bounds__(256)
void normalize_pos(float* __restrict__ praw, const float* __restrict__ red,
                   const float* __restrict__ g, const float* __restrict__ be) {
  const int i = blockIdx.x * 256 + threadIdx.x;
  if (i >= NPTS * DLAT) return;
  const float n   = (float)(NPTS * DLAT);
  const float mu  = red[0] / n;
  const float var = red[1] / n - mu * mu;
  const float rs  = rsqrtf(var + 1e-5f);
  praw[i] = (praw[i] - mu) * rs * g[i] + be[i];
}

// ---------------------------------------------------------------------------
// Permutohedral lattice
// ---------------------------------------------------------------------------
__device__ __forceinline__ unsigned hash64(unsigned long long code) {
  unsigned long long h = code * 0x9E3779B97F4A7C15ull;
  h ^= h >> 32;
  return (unsigned)h & HMASK;
}

__device__ __forceinline__ int hash_find(unsigned long long code,
                                         const unsigned long long* __restrict__ codes,
                                         const int* __restrict__ slots) {
  unsigned h = hash64(code);
  for (unsigned p = 0; p < HSIZE; ++p) {
    const unsigned long long c = codes[h];
    if (c == code)    return slots[h];
    if (c == EMPTY64) return -1;
    h = (h + 1) & HMASK;
  }
  return -1;
}

// Phase 1: per point compute simplex corners, barycentric weights, insert
// codes into hash table (CAS only, no waiting), record slot index + weight.
__global__ __launch_bounds__(256)
void splat_build(const float* __restrict__ pos, unsigned long long* __restrict__ codes,
                 int* __restrict__ cidx, float* __restrict__ cwts) {
  const long long i = (long long)blockIdx.x * 256 + threadIdx.x;

  // elevate: sp = pos*scale ; elevated = sp @ E^T   (d = 8)
  const float SQ23 = 0.8164965809277260f;  // sqrt(2/3)
  float sp[8];
#pragma unroll
  for (int k = 0; k < 8; ++k) {
    const float sc = SQ23 * 9.0f / sqrtf((float)((k + 1) * (k + 2)));
    sp[k] = pos[i * DLAT + k] * sc;
  }
  float suf[NCORN];
  suf[8] = 0.f;
#pragma unroll
  for (int k = 7; k >= 0; --k) suf[k] = sp[k] + suf[k + 1];
  float elev[NCORN];
  elev[0] = suf[0];
#pragma unroll
  for (int r = 1; r < NCORN; ++r) elev[r] = suf[r] - (float)r * sp[r - 1];

  float rem0[NCORN];
  float rsum = 0.f;
#pragma unroll
  for (int r = 0; r < NCORN; ++r) {
    rem0[r] = floorf(elev[r] * (1.f / 9.f) + 0.5f) * 9.f;
    rsum += rem0[r];
  }
  const int sumv = (int)rintf(rsum * (1.f / 9.f));

  float diff[NCORN];
#pragma unroll
  for (int r = 0; r < NCORN; ++r) diff[r] = elev[r] - rem0[r];

  int rk[NCORN];
#pragma unroll
  for (int a = 0; a < NCORN; ++a) {
    int c = 0;
#pragma unroll
    for (int b = 0; b < NCORN; ++b)
      c += (diff[b] > diff[a]) || (diff[b] == diff[a] && b < a);
    rk[a] = c + sumv;
  }
#pragma unroll
  for (int r = 0; r < NCORN; ++r) {
    if (rk[r] < 0)      { rk[r] += NCORN; rem0[r] += 9.f; }
    else if (rk[r] > 8) { rk[r] -= NCORN; rem0[r] -= 9.f; }
  }

  float t[NCORN];
#pragma unroll
  for (int r = 0; r < NCORN; ++r) t[r] = (elev[r] - rem0[r]) * (1.f / 9.f);

  float bary[NCORN + 1];
#pragma unroll
  for (int r = 0; r < NCORN + 1; ++r) bary[r] = 0.f;
#pragma unroll
  for (int r = 0; r < NCORN; ++r) {
    bary[8 - rk[r]] += t[r];
    bary[9 - rk[r]] -= t[r];
  }
  bary[0] += 1.0f + bary[9];

  int ki[8];
#pragma unroll
  for (int k = 0; k < 8; ++k) ki[k] = (int)rem0[k];

#pragma unroll
  for (int c = 0; c < NCORN; ++c) {
    unsigned long long code = 0ull;
#pragma unroll
    for (int k = 0; k < 8; ++k) {
      // canonical[c][q] = (q < 9-c) ? c : c-9
      const int key = ki[k] + ((rk[k] < NCORN - c) ? c : c - NCORN);
      code += (unsigned long long)(long long)key * kMults[k];
    }
    // CAS-insert code; no index assignment here (no spin-wait -> no deadlock)
    unsigned h = hash64(code);
    for (;;) {
      const unsigned long long prev = atomicCAS(&codes[h], EMPTY64, code);
      if (prev == EMPTY64 || prev == code) break;
      h = (h + 1) & HMASK;
    }
    cidx[i * NCORN + c] = (int)h;   // slot for now; remapped in phase 3
    cwts[i * NCORN + c] = bary[c];
  }
}

// Phase 2: compact occupied slots into dense entry ids.
__global__ __launch_bounds__(256)
void enumerate_slots(const unsigned long long* __restrict__ codes,
                     int* __restrict__ slots, int* __restrict__ counter,
                     unsigned long long* __restrict__ ecodes) {
  const unsigned h = blockIdx.x * 256 + threadIdx.x;
  const unsigned long long c = codes[h];
  if (c != EMPTY64) {
    const int e = atomicAdd(counter, 1);
    slots[h]  = e;
    ecodes[e] = c;
  }
}

// Phase 3: remap per-corner slot -> dense entry id.
__global__ __launch_bounds__(256)
void remap_idx(int* __restrict__ cidx, const int* __restrict__ slots) {
  const int m = blockIdx.x * 256 + threadIdx.x;
  if (m < NPTS * NCORN) cidx[m] = slots[cidx[m]];
}

// Splat values: one 64-thread block per (point, corner), atomic add 64 channels.
__global__ __launch_bounds__(64)
void splat_accum(const float* __restrict__ x, const int* __restrict__ cidx,
                 const float* __restrict__ cwts, float* __restrict__ lat0) {
  const int m  = blockIdx.x;      // point*9 + corner
  const int i  = m / NCORN;
  const int e  = cidx[m];
  const float w = cwts[m];
  const int ch = threadIdx.x;
  atomicAdd(&lat0[(long long)e * DDIM + ch], w * x[(long long)i * DDIM + ch]);
}

// One blur pass along lattice direction j: dst = 0.5*src + 0.25*(n+ + n-).
// Neighbor codes are code +/- offc (hash code is linear in the integer key).
__global__ __launch_bounds__(64)
void blur_pass(const float* __restrict__ src, float* __restrict__ dst,
               const unsigned long long* __restrict__ ecodes,
               const unsigned long long* __restrict__ codes,
               const int* __restrict__ slots, const int* __restrict__ counter,
               unsigned long long offc) {
  const int e = blockIdx.x;
  if (e >= *counter) return;
  __shared__ int nb[2];
  if (threadIdx.x < 2) {
    const unsigned long long c = ecodes[e];
    const unsigned long long q = threadIdx.x == 0 ? (c + offc) : (c - offc);
    nb[threadIdx.x] = hash_find(q, codes, slots);
  }
  __syncthreads();
  const int ch = threadIdx.x;
  float v = 0.5f * src[(long long)e * DDIM + ch];
  if (nb[0] >= 0) v += 0.25f * src[(long long)nb[0] * DDIM + ch];
  if (nb[1] >= 0) v += 0.25f * src[(long long)nb[1] * DDIM + ch];
  dst[(long long)e * DDIM + ch] = v;
}

// Slice: out[i,ch] = alpha * sum_c w[i,c] * lat[idx[i,c], ch]
__global__ __launch_bounds__(64)
void slice_k(const int* __restrict__ cidx, const float* __restrict__ cwts,
             const float* __restrict__ lat, float* __restrict__ out) {
  const int i  = blockIdx.x;
  const int ch = threadIdx.x;
  const float alpha = 256.0f / 257.0f;   // 1/(1+2^-8)
  float acc = 0.f;
#pragma unroll
  for (int c = 0; c < NCORN; ++c) {
    const int e   = cidx[i * NCORN + c];
    const float w = cwts[i * NCORN + c];
    acc += w * lat[(long long)e * DDIM + ch];
  }
  out[(long long)i * DDIM + ch] = acc * alpha;
}

// ---------------------------------------------------------------------------
// Host launcher
// ---------------------------------------------------------------------------
extern "C" void kernel_launch(void* const* d_in, const int* in_sizes, int n_in,
                              void* d_out, int out_size, void* d_ws, size_t ws_size,
                              hipStream_t stream) {
  (void)in_sizes; (void)n_in; (void)out_size; (void)ws_size;

  const float* kpts  = (const float*)d_in[0];
  const float* disps = (const float*)d_in[1];
  const float* feats = (const float*)d_in[2];

  // Workspace carve-up (256B aligned)
  char* base = (char*)d_ws;
  size_t off = 0;
  auto carve = [&](size_t bytes) -> char* {
    char* p = base + off;
    off = (off + bytes + 255) & ~(size_t)255;
    return p;
  };
  float*              actA    = (float*)carve((size_t)NPTS * DDIM * 4);
  float*              actB    = (float*)carve((size_t)NPTS * DDIM * 4);
  float*              praw    = (float*)carve((size_t)NPTS * DLAT * 4);
  float*              red     = (float*)carve(16);
  int*                counter = (int*)carve(16);
  unsigned long long* codes   = (unsigned long long*)carve((size_t)HSIZE * 8);
  int*                slots   = (int*)carve((size_t)HSIZE * 4);
  unsigned long long* ecodes  = (unsigned long long*)carve((size_t)MCAP * 8);
  int*                cidx    = (int*)carve((size_t)NPTS * NCORN * 4);
  float*              cwts    = (float*)carve((size_t)NPTS * NCORN * 4);
  float*              lat0    = (float*)carve((size_t)MCAP * DDIM * 4);
  float*              lat1    = (float*)carve((size_t)MCAP * DDIM * 4);

  // blur direction code offsets: o = ones(9), o[j] = -8; OFFC[j] = sum(o[:8]*mults)
  unsigned long long S = 0;
  for (int k = 0; k < 8; ++k) S += kMults[k];
  unsigned long long offc[NCORN];
  for (int j = 0; j < 8; ++j) offc[j] = S - 9ull * kMults[j];
  offc[8] = S;

  for (int b = 0; b < 2; ++b) {
    const float* Wd0 = (const float*)d_in[3 + 8 * b + 0];
    const float* bd0 = (const float*)d_in[3 + 8 * b + 1];
    const float* Wd1 = (const float*)d_in[3 + 8 * b + 2];
    const float* bd1 = (const float*)d_in[3 + 8 * b + 3];
    const float* Wf  = (const float*)d_in[3 + 8 * b + 4];
    const float* bf  = (const float*)d_in[3 + 8 * b + 5];
    const float* g   = (const float*)d_in[3 + 8 * b + 6];
    const float* be  = (const float*)d_in[3 + 8 * b + 7];

    // reset per-block state (all stream ops: graph-capture safe)
    hipMemsetAsync(red, 0, 16, stream);
    hipMemsetAsync(counter, 0, 16, stream);
    hipMemsetAsync(codes, 0xFF, (size_t)HSIZE * 8, stream);
    hipMemsetAsync(slots, 0xFF, (size_t)HSIZE * 4, stream);
    hipMemsetAsync(lat0, 0, (size_t)MCAP * DDIM * 4, stream);

    const float* Xin = (b == 0) ? disps : actB;
    fused_mlp_wmma<<<256, 256, 0, stream>>>(Xin, Wd0, bd0, Wd1, bd1, actA);

    pos_gemm_reduce<<<NPTS / 256, 256, 0, stream>>>(kpts, feats, Wf, bf, praw, red);
    normalize_pos<<<(NPTS * DLAT) / 256, 256, 0, stream>>>(praw, red, g, be);

    splat_build<<<NPTS / 256, 256, 0, stream>>>(praw, codes, cidx, cwts);
    enumerate_slots<<<HSIZE / 256, 256, 0, stream>>>(codes, slots, counter, ecodes);
    remap_idx<<<(NPTS * NCORN + 255) / 256, 256, 0, stream>>>(cidx, slots);
    splat_accum<<<NPTS * NCORN, 64, 0, stream>>>(actA, cidx, cwts, lat0);

    for (int j = 0; j < NCORN; ++j) {
      const float* src = (j & 1) ? lat1 : lat0;
      float*       dst = (j & 1) ? lat0 : lat1;
      blur_pass<<<MCAP, 64, 0, stream>>>(src, dst, ecodes, codes, slots, counter, offc[j]);
    }
    // 9 passes end with data in lat1 (last pass j=8 writes lat1)
    float* outPtr = (b == 0) ? actB : (float*)d_out;
    slice_k<<<NPTS, 64, 0, stream>>>(cidx, cwts, lat1, outPtr);
  }
}
